// GUEST_32615981646067
// MI455X (gfx1250) — matrix operand order repaired and checked by reading, not compiled, not run on hardware
//
#include <hip/hip_runtime.h>
#include <hip/hip_bf16.h>
#include <math.h>

// ---------------------------------------------------------------------------
// CDNA5 (gfx1250) implementation. Matrix-heavy parts use
// v_wmma_f32_16x16x32_f16 (wave32 WMMA) with pre-packed B fragments so the
// GEMM inner loops are pure b128 loads + WMMA. Everything else is f32 VALU.
// ---------------------------------------------------------------------------

typedef __attribute__((ext_vector_type(16))) _Float16 v16h;
typedef __attribute__((ext_vector_type(8)))  _Float16 v8h;
typedef __attribute__((ext_vector_type(8)))  float    v8f;

#define NB   64    // batch
#define NI   64    // NINS
#define NF   256   // NFEAT
#define NH   64    // NHID
#define NE1  64    // NEMBD1
#define NDD  320   // D = NFEAT + NEMBD1
#define LDFS 1280  // 4*D
#define LDH  640   // 2*D

// ---- WMMA fragment loaders -------------------------------------------------
// A-matrix 16x32 f16 fragment in this file's contiguous k order: lane L holds row m=L&15;
// elements 0..15 -> K = kblk*32 + (L>>4)*16 + e (matches packed B).
// 16 contiguous halfs in memory -> two 16-byte loads.
__device__ __forceinline__ v16h load_a_frag(const _Float16* tile, int ld, int kblk) {
  int lane = threadIdx.x & 31;
  int m  = lane & 15;
  int k0 = kblk * 32 + (lane >> 4) * 16;
  const _Float16* p = tile + (size_t)m * ld + k0;
  v8h lo = *(const v8h*)(p);
  v8h hi = *(const v8h*)(p + 8);
  return __builtin_shufflevector(lo, hi, 0,1,2,3,4,5,6,7,8,9,10,11,12,13,14,15);
}

// B fragment from pre-packed layout: tile blob is 32 lanes x 16 halfs,
// lane-major. Per lane: one contiguous 32B chunk -> two b128 loads.
__device__ __forceinline__ v16h load_b_packed(const _Float16* tile) {
  int lane = threadIdx.x & 31;
  const _Float16* p = tile + (size_t)lane * 16;
  v8h lo = *(const v8h*)(p);
  v8h hi = *(const v8h*)(p + 8);
  return __builtin_shufflevector(lo, hi, 0,1,2,3,4,5,6,7,8,9,10,11,12,13,14,15);
}

__device__ __forceinline__ v8f wmma16(v16h a, v16h b, v8f c) {
  return __builtin_amdgcn_wmma_f32_16x16x32_f16(false, a, false, b, (short)0, c,
                                                false, false);
}

// ---- f32 -> f16 convert ----------------------------------------------------
__global__ void __launch_bounds__(256) cvt_kernel(const float* __restrict__ src,
                                                  _Float16* __restrict__ dst, int n) {
  int idx = blockIdx.x * 256 + threadIdx.x;
  if (idx < n) dst[idx] = (_Float16)src[idx];
}

// ---- pack W1_oth (l,i,256..511,64) f32 -> WMMA B-fragment layout f16 -------
// packed idx = (((li*4 + nt)*8 + kblk)*32 + lane)*16 + e
// source element: K = kblk*32 + (lane>>4)*16 + e (row within oth), n = nt*16+(lane&15)
__global__ void __launch_bounds__(256) pack_w1_kernel(const float* __restrict__ W1,
                                                      _Float16* __restrict__ out) {
  int idx = blockIdx.x * 256 + threadIdx.x;  // 0 .. 2,097,151 (exact grid)
  int e    = idx & 15;
  int lane = (idx >> 4) & 31;
  int kblk = (idx >> 9) & 7;
  int nt   = (idx >> 12) & 3;
  int li   = idx >> 14;                      // 0..127
  int K = kblk * 32 + (lane >> 4) * 16 + e;
  int n = nt * 16 + (lane & 15);
  out[idx] = (_Float16)W1[((size_t)li * 512 + 256 + K) * NH + n];
}

// ---- pack entail_W (1280,640) f32 -> WMMA B-fragment layout f16 ------------
// packed idx = ((nt*40 + kblk)*32 + lane)*16 + e, nt 0..39, kblk 0..39
__global__ void __launch_bounds__(256) pack_ew_kernel(const float* __restrict__ W,
                                                      _Float16* __restrict__ out) {
  int idx = blockIdx.x * 256 + threadIdx.x;  // 0 .. 819,199 (exact grid)
  int e    = idx & 15;
  int lane = (idx >> 4) & 31;
  int rest = idx >> 9;                       // 0..1599
  int kblk = rest % 40;
  int nt   = rest / 40;
  int K = kblk * 32 + (lane >> 4) * 16 + e;
  int n = nt * 16 + (lane & 15);
  out[idx] = (_Float16)W[(size_t)K * LDH + n];
}

// ---- GNN attention layer ---------------------------------------------------
// One block per (b,i).  4 waves; wave w owns M-tile w of the 64x64 oth GEMM.
__global__ void __launch_bounds__(128) attn_kernel(
    const float*    __restrict__ X32,   // (B,NI,NF) current inputs, f32
    const _Float16* __restrict__ Xh,    // same, f16
    const float*    __restrict__ W1,    // (L,NI,512,NH) f32 (own projection)
    const _Float16* __restrict__ W1p,   // packed oth B fragments, f16
    const float*    __restrict__ b1,    // (L,NI,NH)
    const float*    __restrict__ w2,    // (L,NI,NH)
    const float*    __restrict__ b2,    // (L,NI)
    float*          __restrict__ Xout,  // (B,NI,NF) f32
    _Float16*       __restrict__ Xouth, // (B,NI,NF) f16 or nullptr
    float*          __restrict__ attn_out, // (B,NI,NI) or nullptr
    int l)
{
  __shared__ float O[64][65];
  __shared__ float ownh[64];
  __shared__ float xi[NF];
  __shared__ float sc[64];
  __shared__ float red[64];
  __shared__ float wsh[64];

  int blk  = blockIdx.x;
  int b    = blk >> 6;
  int i    = blk & 63;
  int t    = threadIdx.x;
  int wave = t >> 5;
  int lane = t & 31;
  int li   = l * NI + i;

  // stage row i (f32) for the "own" projection
  for (int f = t; f < NF; f += 128)
    xi[f] = X32[((size_t)b * NI + i) * NF + f];

  // O[j,h] = Xh[b,j,:] (64x256) @ W1_oth[l,i] (256x64), f16 WMMA, f32 acc
  const _Float16* Ab = Xh + (size_t)b * NI * NF + (size_t)wave * 16 * NF;
  const _Float16* Bp = W1p + (size_t)li * 4 * 8 * 512;  // [nt][kblk][lane][e]
  v8f acc[4];
#pragma unroll
  for (int nt = 0; nt < 4; ++nt)
#pragma unroll
    for (int r = 0; r < 8; ++r) acc[nt][r] = 0.0f;

  for (int k = 0; k < NF / 32; ++k) {
    v16h a = load_a_frag(Ab, NF, k);
#pragma unroll
    for (int nt = 0; nt < 4; ++nt) {
      v16h bf = load_b_packed(Bp + ((size_t)(nt * 8 + k)) * 512);
      acc[nt] = wmma16(a, bf, acc[nt]);
    }
  }
  int lh = lane >> 4, ln = lane & 15;
#pragma unroll
  for (int nt = 0; nt < 4; ++nt)
#pragma unroll
    for (int r = 0; r < 8; ++r)
      O[wave * 16 + lh * 8 + r][nt * 16 + ln] = acc[nt][r];

  __syncthreads();

  // own[h] = xi . W1_own[l,i][:,h] + b1[l,i,h]
  if (t < NH) {
    float s = b1[(size_t)li * NH + t];
    const float* wp = W1 + (size_t)li * 512 * NH + t;
    for (int f = 0; f < NF; ++f) s += xi[f] * wp[(size_t)f * NH];
    ownh[t] = s;
  }
  __syncthreads();

  // scores[j] = leaky_relu(O[j,h] + own[h], 0.1) . w2[l,i] + b2[l,i]
  if (t < NI) {
    float s = b2[li];
    const float* w2p = w2 + (size_t)li * NH;
    for (int h = 0; h < NH; ++h) {
      float v = O[t][h] + ownh[h];
      v = v > 0.0f ? v : 0.1f * v;
      s += v * w2p[h];
    }
    sc[t] = s;
  }
  __syncthreads();

  // softmax over j (64)
  if (t < 64) red[t] = sc[t];
  __syncthreads();
  for (int s = 32; s > 0; s >>= 1) {
    if (t < s) red[t] = fmaxf(red[t], red[t + s]);
    __syncthreads();
  }
  float mx = red[0];
  __syncthreads();
  if (t < 64) { float e = __expf(sc[t] - mx); sc[t] = e; red[t] = e; }
  __syncthreads();
  for (int s = 32; s > 0; s >>= 1) {
    if (t < s) red[t] += red[t + s];
    __syncthreads();
  }
  float inv = 1.0f / red[0];
  __syncthreads();
  if (t < 64) wsh[t] = sc[t] * inv;
  __syncthreads();

  // attns4visual = swapaxes(w,1,2): out[b][j][i] = w[b][i][j]
  if (attn_out != nullptr && t < 64)
    attn_out[((size_t)b * NI + t) * NI + i] = wsh[t];

  // new_inputs[b,i,f] = sum_j w[j] * X32[b,j,f]   (also emit f16 copy)
  for (int f = t; f < NF; f += 128) {
    float s = 0.0f;
    const float* xp = X32 + (size_t)b * NI * NF + f;
    for (int j = 0; j < NI; ++j) s += wsh[j] * xp[(size_t)j * NF];
    size_t oidx = ((size_t)b * NI + i) * NF + f;
    Xout[oidx] = s;
    if (Xouth != nullptr) Xouth[oidx] = (_Float16)s;
  }
}

// ---- build fs features (f16) + bc output ----------------------------------
__global__ void __launch_bounds__(320) fs_build_kernel(
    const float* __restrict__ Xfin,   // (B,NI,NF) final GNN output
    const float* __restrict__ claims, // (B,1,NF)
    const float* __restrict__ bcred,  // (B,NI+1)
    const float* __restrict__ fignn,  // (B,NI+1,NE1)
    _Float16*    __restrict__ FSh,    // (B,NI,4D) f16
    float*       __restrict__ out_bc) // (B,NI)
{
  int blk = blockIdx.x;
  int b = blk >> 6;
  int n = blk & 63;
  int d = threadIdx.x;  // 0..319
  float bc = bcred[(size_t)b * (NI + 1) + 1 + n];
  float c, cm;
  if (d < NF) {
    c  = claims[(size_t)b * NF + d];
    cm = Xfin[((size_t)b * NI + n) * NF + d];
  } else {
    c  = fignn[((size_t)b * (NI + 1) + 0) * NE1 + (d - NF)];
    cm = bc * fignn[((size_t)b * (NI + 1) + 1 + n) * NE1 + (d - NF)];
  }
  _Float16* o = FSh + ((size_t)b * NI + n) * LDFS;
  o[d]           = (_Float16)c;
  o[NDD + d]     = (_Float16)cm;
  o[2 * NDD + d] = (_Float16)(c * cm);
  o[3 * NDD + d] = (_Float16)fabsf(c - cm);
  if (d == 0) out_bc[(size_t)b * NI + n] = bc;
}

// ---- entailment GEMM: hent = tanh(fs @ entail_W + entail_b) ----------------
// Grid: B*10 blocks; block covers 4 N-tiles (of 40), 4 waves = 4 M-tiles.
__global__ void __launch_bounds__(128) entail_kernel(
    const _Float16* __restrict__ FSh,  // (B,NI,1280)
    const _Float16* __restrict__ EWp,  // packed B fragments [nt][kblk][lane][e]
    const float*    __restrict__ eb,   // (640)
    float*          __restrict__ HENT) // (B,NI,640)
{
  int blk  = blockIdx.x;
  int b    = blk / 10;
  int nb   = blk % 10;
  int wave = threadIdx.x >> 5;
  int lane = threadIdx.x & 31;
  const _Float16* Ab = FSh + (size_t)b * NI * LDFS + (size_t)wave * 16 * LDFS;
  v8f acc[4];
#pragma unroll
  for (int nt = 0; nt < 4; ++nt)
#pragma unroll
    for (int r = 0; r < 8; ++r) acc[nt][r] = 0.0f;

  for (int k = 0; k < LDFS / 32; ++k) {
    v16h a = load_a_frag(Ab, LDFS, k);
#pragma unroll
    for (int nt = 0; nt < 4; ++nt) {
      int ntg = nb * 4 + nt;
      v16h bf = load_b_packed(EWp + ((size_t)(ntg * 40 + k)) * 512);
      acc[nt] = wmma16(a, bf, acc[nt]);
    }
  }
  int lh = lane >> 4, ln = lane & 15;
#pragma unroll
  for (int nt = 0; nt < 4; ++nt) {
    int col = (nb * 4 + nt) * 16 + ln;
    float bias = eb[col];
#pragma unroll
    for (int r = 0; r < 8; ++r) {
      int row = wave * 16 + lh * 8 + r;
      HENT[((size_t)b * NI + row) * LDH + col] = tanhf(acc[nt][r] + bias);
    }
  }
}

// ---- entail2 + softmax + weighted sum + FFN + log_softmax ------------------
__global__ void __launch_bounds__(256) final_kernel(
    const float* __restrict__ HENT, const float* __restrict__ e2W,
    const float* __restrict__ e2b,  const float* __restrict__ f1W,
    const float* __restrict__ f1b,  const float* __restrict__ f2W,
    const float* __restrict__ f2b,  float* __restrict__ out_log,
    float* __restrict__ out_fsW)
{
  __shared__ float sc[64];
  __shared__ float red[64];
  __shared__ float fw[64];
  __shared__ float ff[LDH];
  __shared__ float y1[NDD];
  __shared__ float o2[2];
  int b = blockIdx.x;
  int t = threadIdx.x;

  if (t < 64) {
    float s = e2b[0];
    const float* hp = HENT + ((size_t)b * NI + t) * LDH;
    for (int h = 0; h < LDH; ++h) s += hp[h] * e2W[h];
    sc[t] = tanhf(s);
  }
  __syncthreads();
  if (t < 64) red[t] = sc[t];
  __syncthreads();
  for (int s = 32; s > 0; s >>= 1) {
    if (t < s) red[t] = fmaxf(red[t], red[t + s]);
    __syncthreads();
  }
  float mx = red[0];
  __syncthreads();
  if (t < 64) { float e = __expf(sc[t] - mx); sc[t] = e; red[t] = e; }
  __syncthreads();
  for (int s = 32; s > 0; s >>= 1) {
    if (t < s) red[t] += red[t + s];
    __syncthreads();
  }
  float inv = 1.0f / red[0];
  __syncthreads();
  if (t < 64) { fw[t] = sc[t] * inv; out_fsW[(size_t)b * NI + t] = fw[t]; }
  __syncthreads();

  for (int h = t; h < LDH; h += 256) {
    float s = 0.0f;
    for (int n = 0; n < NI; ++n) s += fw[n] * HENT[((size_t)b * NI + n) * LDH + h];
    ff[h] = s;
  }
  __syncthreads();
  for (int k = t; k < NDD; k += 256) {
    float s = f1b[k];
    for (int h = 0; h < LDH; ++h) s += ff[h] * f1W[(size_t)h * NDD + k];
    y1[k] = s;
  }
  __syncthreads();
  if (t < 2) {
    float s = f2b[t];
    for (int k = 0; k < NDD; ++k) s += y1[k] * f2W[(size_t)k * 2 + t];
    o2[t] = s;
  }
  __syncthreads();
  if (t == 0) {
    float m = fmaxf(o2[0], o2[1]);
    float lse = m + logf(__expf(o2[0] - m) + __expf(o2[1] - m));
    out_log[(size_t)b * 2 + 0] = o2[0] - lse;
    out_log[(size_t)b * 2 + 1] = o2[1] - lse;
  }
}

// ---------------------------------------------------------------------------
extern "C" void kernel_launch(void* const* d_in, const int* in_sizes, int n_in,
                              void* d_out, int out_size, void* d_ws, size_t ws_size,
                              hipStream_t stream) {
  (void)in_sizes; (void)n_in; (void)out_size; (void)ws_size;
  const float* inputs = (const float*)d_in[0];
  const float* claims = (const float*)d_in[1];
  const float* bcred  = (const float*)d_in[2];
  const float* fignn  = (const float*)d_in[3];
  const float* attnW1 = (const float*)d_in[4];
  const float* attnb1 = (const float*)d_in[5];
  const float* attnw2 = (const float*)d_in[6];
  const float* attnb2 = (const float*)d_in[7];
  const float* entW   = (const float*)d_in[8];
  const float* entb   = (const float*)d_in[9];
  const float* ent2W  = (const float*)d_in[10];
  const float* ent2b  = (const float*)d_in[11];
  const float* f1W    = (const float*)d_in[12];
  const float* f1b    = (const float*)d_in[13];
  const float* f2W    = (const float*)d_in[14];
  const float* f2b    = (const float*)d_in[15];

  // workspace layout (all offsets 256B aligned)
  char* ws = (char*)d_ws;
  _Float16* W1p  = (_Float16*)(ws + 0);         //  4,194,304 B (packed oth B)
  _Float16* EWp  = (_Float16*)(ws + 4194304);   //  1,638,400 B (packed entail B)
  _Float16* Xh   = (_Float16*)(ws + 5832704);   //  2,097,152 B
  _Float16* Xh2  = (_Float16*)(ws + 7929856);   //  2,097,152 B
  float*    Xb1  = (float*)   (ws + 10027008);  //  4,194,304 B
  float*    Xb2  = (float*)   (ws + 14221312);  //  4,194,304 B
  _Float16* FSh  = (_Float16*)(ws + 18415616);  // 10,485,760 B
  float*    HENT = (float*)   (ws + 28901376);  // 10,485,760 B  (total ~39.4 MB)

  float* out      = (float*)d_out;
  float* out_log  = out;                          // (64,2)
  float* out_attn = out + 128;                    // (64,64,64)
  float* out_bc   = out + 128 + 262144;           // (64,64)
  float* out_fsW  = out + 128 + 262144 + 4096;    // (64,64,1)

  const int NW1P = 2 * NI * 4 * 8 * 512;  // 2,097,152 packed halfs
  const int NEWP = 40 * 40 * 512;         //   819,200 packed halfs
  const int NX   = NB * NI * NF;          // 1,048,576

  pack_w1_kernel<<<NW1P / 256, 256, 0, stream>>>(attnW1, W1p);
  pack_ew_kernel<<<NEWP / 256, 256, 0, stream>>>(entW, EWp);
  cvt_kernel<<<(NX + 255) / 256, 256, 0, stream>>>(inputs, Xh, NX);

  attn_kernel<<<NB * NI, 128, 0, stream>>>(inputs, Xh, attnW1, W1p, attnb1,
                                           attnw2, attnb2, Xb1, Xh2, nullptr, 0);
  attn_kernel<<<NB * NI, 128, 0, stream>>>(Xb1, Xh2, attnW1, W1p, attnb1,
                                           attnw2, attnb2, Xb2, nullptr, out_attn, 1);

  fs_build_kernel<<<NB * NI, 320, 0, stream>>>(Xb2, claims, bcred, fignn, FSh, out_bc);
  entail_kernel<<<NB * 10, 128, 0, stream>>>(FSh, EWp, entb, HENT);
  final_kernel<<<NB, 256, 0, stream>>>(HENT, ent2W, ent2b, f1W, f1b, f2W, f2b,
                                       out_log, out_fsW);
}
